// NeuralMemory_81570018886259
// MI455X (gfx1250) — matrix operand order repaired and checked by reading, not compiled, not run on hardware
//
#include <hip/hip_runtime.h>

typedef __attribute__((ext_vector_type(16))) __bf16 v16bf;
typedef __attribute__((ext_vector_type(8)))  __bf16 v8bf;
typedef __attribute__((ext_vector_type(8)))  float  v8f;
typedef __bf16 bf16;

#define DEV static __device__ __forceinline__

namespace {
constexpr int kB = 4, kN = 4096, kDIM = 512, kH = 4, kNC = 64, kBH = 16;
}

DEV int lane_id() { return (int)(threadIdx.x & 31u); }

DEV float wave_sum(float v) {
#pragma unroll
  for (int m = 16; m >= 1; m >>= 1) v += __shfl_xor(v, m, 32);
  return v;
}

DEV float sigmoidf_(float x) { return 1.f / (1.f + __expf(-x)); }
DEV float siluf_(float x)    { return x / (1.f + __expf(-x)); }
DEV float dsiluf_(float x)   { float s = 1.f/(1.f+__expf(-x)); return s*(1.f + x*(1.f - s)); }

DEV v8f wmma_bf16(v16bf a, v16bf b, v8f c) {
  return __builtin_amdgcn_wmma_f32_16x16x32_bf16(false, a, false, b, (short)0, c, false, false);
}

// ---- vectorized WMMA fragment loaders (wave32 layouts, CDNA5 ISA 7.12.2) ----
// A 16x32 bf16 row-major: lane m=L%16, hi=L/16 holds K = [8*hi, 8*hi+8) and
// [16+8*hi, 16+8*hi+8): two contiguous 16B runs -> two ds_load_b128.
DEV v16bf ld_a_rm(const bf16* p, int ld) {   // A[m][k] = p[m*ld + k]
  int l = lane_id(), m = l & 15, hi = l >> 4;
  const bf16* q = p + m * ld + 8 * hi;
  v8bf lo = *(const v8bf*)(q);
  v8bf hi8 = *(const v8bf*)(q + 16);
  return __builtin_shufflevector(lo, hi8, 0,1,2,3,4,5,6,7,8,9,10,11,12,13,14,15);
}
// B 32x16 bf16 from COLUMN-MAJOR storage: B[k][n] = p[n*ldK + k].
// lane n=L%16, hi=L/16 holds K = [8*hi, 8*hi+8) and [16+8*hi, 16+8*hi+8).
DEV v16bf ld_b_ct(const bf16* p, int ldK) {
  int l = lane_id(), n = l & 15, hi = l >> 4;
  const bf16* q = p + n * ldK + 8 * hi;
  v8bf lo = *(const v8bf*)(q);
  v8bf hi8 = *(const v8bf*)(q + 16);
  return __builtin_shufflevector(lo, hi8, 0,1,2,3,4,5,6,7,8,9,10,11,12,13,14,15);
}
// Same, but from a column-major f32 LDS weight state (convert on the fly).
DEV v16bf ld_b_ct_f32(const float* p, int ldK) {
  int l = lane_id(), n = l & 15, hi = l >> 4;
  const float* q = p + n * ldK + 8 * hi;
  float4 f0 = *(const float4*)(q + 0);
  float4 f1 = *(const float4*)(q + 4);
  float4 f2 = *(const float4*)(q + 16);
  float4 f3 = *(const float4*)(q + 20);
  v16bf b;
  b[0]=(bf16)f0.x;  b[1]=(bf16)f0.y;  b[2]=(bf16)f0.z;  b[3]=(bf16)f0.w;
  b[4]=(bf16)f1.x;  b[5]=(bf16)f1.y;  b[6]=(bf16)f1.z;  b[7]=(bf16)f1.w;
  b[8]=(bf16)f2.x;  b[9]=(bf16)f2.y;  b[10]=(bf16)f2.z; b[11]=(bf16)f2.w;
  b[12]=(bf16)f3.x; b[13]=(bf16)f3.y; b[14]=(bf16)f3.z; b[15]=(bf16)f3.w;
  return b;
}

// ---------------------------------------------------------------
// Weight conversion kernels: plain and transposing (dst[c][r] = src[r][c]).
// ---------------------------------------------------------------
__global__ void nm_f32_to_bf16(const float* __restrict__ src, bf16* __restrict__ dst, int n) {
  int i = blockIdx.x * 256 + threadIdx.x;
  if (i < n) dst[i] = (bf16)src[i];
}
__global__ void nm_f32_to_bf16_t(const float* __restrict__ src, bf16* __restrict__ dst,
                                 int rows, int cols) {
  int i = blockIdx.x * 256 + threadIdx.x;
  if (i < rows * cols) {
    int r = i / cols, c = i % cols;
    dst[(size_t)c * rows + r] = (bf16)src[i];
  }
}

// ---------------------------------------------------------------
// Kernel: prep — rmsnorm, g_store/g_retrieve, lr/gate projections,
// chunk means -> decay/mom. One block per (b, chunk); 8 waves, 1 row/wave.
// ---------------------------------------------------------------
__global__ void nm_prep(const float* __restrict__ seq,
                        const float* __restrict__ gs, const float* __restrict__ gr,
                        const float* __restrict__ Wstep, const float* __restrict__ Wgate,
                        const float* __restrict__ Wdecay, const float* __restrict__ Wmom,
                        bf16* __restrict__ s_bf, bf16* __restrict__ r_bf,
                        float* __restrict__ lr, float* __restrict__ gate,
                        float* __restrict__ decay, float* __restrict__ mom) {
  __shared__ float paccum[8 * 512];
  int b = blockIdx.x / kNC, t = blockIdx.x % kNC;
  int tid = threadIdx.x, wv = tid >> 5, l = tid & 31;

  for (int i = tid; i < 8 * 512; i += 256) paccum[i] = 0.f;
  __syncthreads();

  for (int r = wv; r < 64; r += 8) {
    int n = t * 64 + r;
    const float* row = seq + ((size_t)b * kN + n) * kDIM;
    float x[16]; float ss = 0.f;
#pragma unroll
    for (int i = 0; i < 16; ++i) { x[i] = row[l + 32 * i]; ss += x[i] * x[i]; }
    ss = wave_sum(ss);
    float rms = rsqrtf(ss * (1.f / 512.f) + 1e-6f);
    float ps[4] = {0.f, 0.f, 0.f, 0.f}, pg[4] = {0.f, 0.f, 0.f, 0.f};
#pragma unroll
    for (int i = 0; i < 16; ++i) {
      int idx = l + 32 * i;
      float sv = x[i] * rms * gs[idx];
      float rv = x[i] * rms * gr[idx];
      s_bf[((size_t)b * kN + n) * kDIM + idx] = (bf16)sv;
      r_bf[((size_t)b * kN + n) * kDIM + idx] = (bf16)rv;
      paccum[wv * 512 + idx] += sv;
#pragma unroll
      for (int hh = 0; hh < 4; ++hh) {
        ps[hh] += sv * Wstep[idx * 4 + hh];
        pg[hh] += rv * Wgate[idx * 4 + hh];
      }
    }
#pragma unroll
    for (int hh = 0; hh < 4; ++hh) { ps[hh] = wave_sum(ps[hh]); pg[hh] = wave_sum(pg[hh]); }
    if (l == 0) {
#pragma unroll
      for (int hh = 0; hh < 4; ++hh) {
        lr[(((size_t)b * kH + hh) * kNC + t) * 64 + r] = sigmoidf_(ps[hh]); // MAX_LR = 1
        gate[((size_t)b * kN + n) * kH + hh] = sigmoidf_(pg[hh]);
      }
    }
  }
  __syncthreads();

  if (wv == 0) {
    float pd[4] = {0.f, 0.f, 0.f, 0.f}, pm[4] = {0.f, 0.f, 0.f, 0.f};
    for (int idx = l; idx < 512; idx += 32) {
      float mv = 0.f;
#pragma unroll
      for (int w = 0; w < 8; ++w) mv += paccum[w * 512 + idx];
      mv *= (1.f / 64.f);
#pragma unroll
      for (int hh = 0; hh < 4; ++hh) {
        pd[hh] += mv * Wdecay[idx * 4 + hh];
        pm[hh] += mv * Wmom[idx * 4 + hh];
      }
    }
#pragma unroll
    for (int hh = 0; hh < 4; ++hh) { pd[hh] = wave_sum(pd[hh]); pm[hh] = wave_sum(pm[hh]); }
    if (l == 0) {
#pragma unroll
      for (int hh = 0; hh < 4; ++hh) {
        decay[((size_t)b * kH + hh) * kNC + t] = sigmoidf_(pd[hh]);
        mom[((size_t)b * kH + hh) * kNC + t]   = sigmoidf_(pm[hh]);
      }
    }
  }
}

// ---------------------------------------------------------------
// Projection GEMM: Y = X(16384x512)bf16 @ W(512xNout)bf16, W given TRANSPOSED
// (Wt is Nout x 512 row-major), output scattered into chunked bf16 layout.
// ---------------------------------------------------------------
DEV void store_chunked(const v8f& acc, int row0, int col0, bf16* outK, bf16* outV) {
  int l = lane_id(), colL = l & 15, rb = (l >> 4) * 8;
#pragma unroll
  for (int i = 0; i < 8; ++i) {
    int R = row0 + rb + i;
    int Cc = col0 + colL;
    int b = R >> 12, n = R & 4095, t = n >> 6, rc = n & 63;
    bf16* dst = outK; int cc = Cc;
    if (outV != nullptr && Cc >= 512) { dst = outV; cc = Cc - 512; }
    int hh = cc >> 7, d = cc & 127;
    dst[((((size_t)b * kH + hh) * kNC + t) * 64 + rc) * 128 + d] = (bf16)acc[i];
  }
}

__global__ void nm_proj(const bf16* __restrict__ X, const bf16* __restrict__ Wt, int Nout,
                        bf16* __restrict__ outK, bf16* __restrict__ outV) {
  __shared__ bf16 Xs[64 * 64];   // rows x K-panel, row-major
  __shared__ bf16 Wst[64 * 64];  // cols x K-panel (column-major B)
  int tid = threadIdx.x, wv = tid >> 5;
  int row0 = blockIdx.x * 64, col0 = blockIdx.y * 64;
  (void)Nout;
  int tr0 = wv >> 2, tc0 = wv & 3, tr1 = (wv + 8) >> 2, tc1 = (wv + 8) & 3;
  v8f acc0 = {}, acc1 = {};

  for (int kk = 0; kk < 512; kk += 64) {
    for (int i = tid; i < 512; i += 256) {
      int r = i >> 3, c = i & 7;
      ((uint4*)(Xs + r * 64))[c]  = ((const uint4*)(X  + ((size_t)(row0 + r)) * 512 + kk))[c];
      ((uint4*)(Wst + r * 64))[c] = ((const uint4*)(Wt + ((size_t)(col0 + r)) * 512 + kk))[c];
    }
    __syncthreads();
#pragma unroll
    for (int ks = 0; ks < 64; ks += 32) {
      acc0 = wmma_bf16(ld_a_rm(Xs + tr0 * 16 * 64 + ks, 64),
                       ld_b_ct(Wst + tc0 * 16 * 64 + ks, 64), acc0);
      acc1 = wmma_bf16(ld_a_rm(Xs + tr1 * 16 * 64 + ks, 64),
                       ld_b_ct(Wst + tc1 * 16 * 64 + ks, 64), acc1);
    }
    __syncthreads();
  }
  store_chunked(acc0, row0 + tr0 * 16, col0 + tc0 * 16, outK, outV);
  store_chunked(acc1, row0 + tr1 * 16, col0 + tc1 * 16, outK, outV);
}

// ---------------------------------------------------------------
// Per-chunk gradients at base weights (1024 independent blocks).
//   h = k@W1 ; a = silu(h) ; pred = a@W2 ; dp = (2/128)*lr*(pred - v)
//   g2 = a^T@dp ; da = dp@W2^T ; dh = da*silu'(h) ; g1 = k^T@dh
// s1=-g1, s2=-g2 stored COLUMN-MAJOR bf16 (matches scan-state layout).
// ---------------------------------------------------------------
__global__ void nm_grad(const bf16* __restrict__ kbuf, const bf16* __restrict__ vbuf,
                        const bf16* __restrict__ W1t_g, const bf16* __restrict__ W2t_g,
                        const bf16* __restrict__ W2rm_g, const float* __restrict__ lr,
                        bf16* __restrict__ g1buf, bf16* __restrict__ g2buf) {
  extern __shared__ char smem[];
  float* Hcm  = (float*)smem;                      // 64x128 f32, column-major (32KB)
  bf16*  Ktr  = (bf16*)(smem + 32768);             // k tile row-major   (16KB)
  bf16*  Ktc  = Ktr + 64 * 128;                    // k tile col-major   (16KB)
  bf16*  Vt   = Ktc + 64 * 128;                    // v tile row-major   (16KB)
  bf16*  W1t  = Vt + 64 * 128;                     // W1 col-major       (32KB)
  bf16*  W2t  = W1t + 128 * 128;                   // W2 col-major       (32KB)
  bf16*  W2rm = W2t + 128 * 128;                   // W2 row-major       (32KB)
  bf16*  Arm  = W2rm + 128 * 128;                  // a row-major        (16KB)
  bf16*  Acm  = Arm + 64 * 128;                    // a col-major        (16KB)
  bf16*  DPrm = Acm + 64 * 128;                    // dp row-major       (16KB)
  bf16*  DPcm = DPrm + 64 * 128;                   // dp col-major       (16KB)
  bf16*  DHcm = DPcm + 64 * 128;                   // dh col-major       (16KB)

  int t = blockIdx.x, bh = blockIdx.y;
  int tid = threadIdx.x, wv = tid >> 5, l = tid & 31;
  int colL = l & 15, rb = (l >> 4) * 8;

  const bf16* ksrc = kbuf + ((size_t)bh * kNC + t) * 64 * 128;
  const bf16* vsrc = vbuf + ((size_t)bh * kNC + t) * 64 * 128;
  for (int i = tid; i < 1024; i += 256) {          // vectorized row-major copies
    ((uint4*)Ktr)[i] = ((const uint4*)ksrc)[i];
    ((uint4*)Vt)[i]  = ((const uint4*)vsrc)[i];
  }
  for (int i = tid; i < 2048; i += 256) {
    ((uint4*)W1t)[i]  = ((const uint4*)W1t_g)[i];
    ((uint4*)W2t)[i]  = ((const uint4*)W2t_g)[i];
    ((uint4*)W2rm)[i] = ((const uint4*)W2rm_g)[i];
  }
  for (int e = tid; e < 8192; e += 256) {          // k tile transpose (coalesced reads)
    int r = e >> 7, c = e & 127;
    Ktc[c * 64 + r] = ksrc[e];
  }
  __syncthreads();

  // ---- phase 1: H = K @ W1 ; A = silu(H) ----
#pragma unroll
  for (int j = 0; j < 4; ++j) {
    int st = wv * 4 + j, tr = st >> 3, tc = st & 7;
    v8f acc = {};
#pragma unroll
    for (int ks = 0; ks < 128; ks += 32)
      acc = wmma_bf16(ld_a_rm(Ktr + tr * 16 * 128 + ks, 128),
                      ld_b_ct(W1t + tc * 16 * 128 + ks, 128), acc);
    int cc = tc * 16 + colL, r0 = tr * 16 + rb;
    float4 h0, h1; v8bf pa;
    h0.x = acc[0]; h0.y = acc[1]; h0.z = acc[2]; h0.w = acc[3];
    h1.x = acc[4]; h1.y = acc[5]; h1.z = acc[6]; h1.w = acc[7];
    *(float4*)(Hcm + cc * 64 + r0) = h0;
    *(float4*)(Hcm + cc * 64 + r0 + 4) = h1;
#pragma unroll
    for (int i = 0; i < 8; ++i) {
      float a = siluf_(acc[i]);
      pa[i] = (bf16)a;
      Arm[(r0 + i) * 128 + cc] = (bf16)a;
    }
    *(v8bf*)(Acm + cc * 64 + r0) = pa;
  }
  __syncthreads();

  // ---- phase 2: PRED = A @ W2 ; DP = (2/128)*lr*(pred - v) ----
  const float* lrc = lr + ((size_t)bh * kNC + t) * 64;
#pragma unroll
  for (int j = 0; j < 4; ++j) {
    int st = wv * 4 + j, tr = st >> 3, tc = st & 7;
    v8f acc = {};
#pragma unroll
    for (int ks = 0; ks < 128; ks += 32)
      acc = wmma_bf16(ld_a_rm(Arm + tr * 16 * 128 + ks, 128),
                      ld_b_ct(W2t + tc * 16 * 128 + ks, 128), acc);
    int cc = tc * 16 + colL, r0 = tr * 16 + rb;
    v8bf pdp;
#pragma unroll
    for (int i = 0; i < 8; ++i) {
      int rr = r0 + i;
      float dp = (acc[i] - (float)Vt[rr * 128 + cc]) * lrc[rr] * (2.f / 128.f);
      pdp[i] = (bf16)dp;
      DPrm[rr * 128 + cc] = (bf16)dp;
    }
    *(v8bf*)(DPcm + cc * 64 + r0) = pdp;
  }
  __syncthreads();

  // ---- phase 3: G2 = A^T @ DP -> s2 = -G2 (column-major global) ----
  bf16* g2dst = g2buf + ((size_t)bh * kNC + t) * 128 * 128;
#pragma unroll
  for (int j = 0; j < 8; ++j) {
    int st = wv * 8 + j, tr = st >> 3, tc = st & 7;
    v8f acc = {};
#pragma unroll
    for (int ks = 0; ks < 64; ks += 32)
      acc = wmma_bf16(ld_a_rm(Acm + tr * 16 * 64 + ks, 64),   // (a^T) row-major
                      ld_b_ct(DPcm + tc * 16 * 64 + ks, 64), acc);
    int cc = tc * 16 + colL, r0 = tr * 16 + rb;
    v8bf pk;
#pragma unroll
    for (int i = 0; i < 8; ++i) pk[i] = (bf16)(-acc[i]);
    *(v8bf*)(g2dst + (size_t)cc * 128 + r0) = pk;
  }

  // ---- phase 4: DA = DP @ W2^T ; DH = DA * silu'(H) ----
#pragma unroll
  for (int j = 0; j < 4; ++j) {
    int st = wv * 4 + j, tr = st >> 3, tc = st & 7;
    v8f acc = {};
#pragma unroll
    for (int ks = 0; ks < 128; ks += 32)
      acc = wmma_bf16(ld_a_rm(DPrm + tr * 16 * 128 + ks, 128),
                      ld_b_ct(W2rm + tc * 16 * 128 + ks, 128), acc); // W2^T col-major == W2 row-major
    int cc = tc * 16 + colL, r0 = tr * 16 + rb;
    float4 h0 = *(const float4*)(Hcm + cc * 64 + r0);
    float4 h1 = *(const float4*)(Hcm + cc * 64 + r0 + 4);
    float hv[8] = {h0.x, h0.y, h0.z, h0.w, h1.x, h1.y, h1.z, h1.w};
    v8bf pk;
#pragma unroll
    for (int i = 0; i < 8; ++i) pk[i] = (bf16)(acc[i] * dsiluf_(hv[i]));
    *(v8bf*)(DHcm + cc * 64 + r0) = pk;
  }
  __syncthreads();

  // ---- phase 5: G1 = K^T @ DH -> s1 = -G1 (column-major global) ----
  bf16* g1dst = g1buf + ((size_t)bh * kNC + t) * 128 * 128;
#pragma unroll
  for (int j = 0; j < 8; ++j) {
    int st = wv * 8 + j, tr = st >> 3, tc = st & 7;
    v8f acc = {};
#pragma unroll
    for (int ks = 0; ks < 64; ks += 32)
      acc = wmma_bf16(ld_a_rm(Ktc + tr * 16 * 64 + ks, 64),   // (k^T) row-major
                      ld_b_ct(DHcm + tc * 16 * 64 + ks, 64), acc);
    int cc = tc * 16 + colL, r0 = tr * 16 + rb;
    v8bf pk;
#pragma unroll
    for (int i = 0; i < 8; ++i) pk[i] = (bf16)(-acc[i]);
    *(v8bf*)(g1dst + (size_t)cc * 128 + r0) = pk;
  }
}

// ---------------------------------------------------------------
// Sequential scan + retrieval. One block per (b*H+h). W1/W2/m1/m2 are f32
// COLUMN-MAJOR LDS state (256KB) + q/a tiles (32KB) = 288KB of 320KB LDS.
// Per chunk: retrieve with current W, then m = mom*m + s_t, W = (1-decay)*W + m.
// ---------------------------------------------------------------
__global__ void nm_scan(const float* __restrict__ MW1, const float* __restrict__ MW2,
                        const bf16* __restrict__ qbuf,
                        const bf16* __restrict__ g1buf, const bf16* __restrict__ g2buf,
                        const float* __restrict__ mom, const float* __restrict__ decay,
                        float* __restrict__ predbuf) {
  extern __shared__ char smem[];
  float* W1f = (float*)smem;                       // col-major, 64KB
  float* W2f = W1f + 128 * 128;
  float* M1f = W2f + 128 * 128;
  float* M2f = M1f + 128 * 128;
  bf16*  Qbf = (bf16*)(M2f + 128 * 128);           // 16KB row-major
  bf16*  Abf = Qbf + 64 * 128;                     // 16KB row-major

  int bh = blockIdx.x;
  int tid = threadIdx.x, wv = tid >> 5, l = tid & 31;
  int colL = l & 15, rb = (l >> 4) * 8;

  for (int i = tid; i < 16384; i += 256) {         // transposed init (col-major state)
    W1f[i] = MW1[(i & 127) * 128 + (i >> 7)];
    W2f[i] = MW2[(i & 127) * 128 + (i >> 7)];
    M1f[i] = 0.f; M2f[i] = 0.f;
  }
  __syncthreads();

  for (int t = 0; t < kNC; ++t) {
    const bf16* qsrc = qbuf + ((size_t)bh * kNC + t) * 64 * 128;
    for (int i = tid; i < 1024; i += 256) ((uint4*)Qbf)[i] = ((const uint4*)qsrc)[i];
    if (t + 1 < kNC) {   // prefetch next chunk's gradients behind the matrix work
      __builtin_prefetch(g1buf + ((size_t)bh * kNC + (t + 1)) * 16384 + tid * 64, 0, 1);
      __builtin_prefetch(g2buf + ((size_t)bh * kNC + (t + 1)) * 16384 + tid * 64, 0, 1);
    }
    __syncthreads();

    // Hq = Q @ W1 ; A = silu(Hq)
#pragma unroll
    for (int j = 0; j < 4; ++j) {
      int st = wv * 4 + j, tr = st >> 3, tc = st & 7;
      v8f acc = {};
#pragma unroll
      for (int ks = 0; ks < 128; ks += 32)
        acc = wmma_bf16(ld_a_rm(Qbf + tr * 16 * 128 + ks, 128),
                        ld_b_ct_f32(W1f + tc * 16 * 128 + ks, 128), acc);
      int cc = tc * 16 + colL, r0 = tr * 16 + rb;
#pragma unroll
      for (int i = 0; i < 8; ++i) Abf[(r0 + i) * 128 + cc] = (bf16)siluf_(acc[i]);
    }
    __syncthreads();

    // PRED = A @ W2 -> predbuf[(b*H+h), n = t*64+row, d]
    float* dst = predbuf + ((size_t)bh * kN + (size_t)t * 64) * 128;
#pragma unroll
    for (int j = 0; j < 4; ++j) {
      int st = wv * 4 + j, tr = st >> 3, tc = st & 7;
      v8f acc = {};
#pragma unroll
      for (int ks = 0; ks < 128; ks += 32)
        acc = wmma_bf16(ld_a_rm(Abf + tr * 16 * 128 + ks, 128),
                        ld_b_ct_f32(W2f + tc * 16 * 128 + ks, 128), acc);
      int cc = tc * 16 + colL, r0 = tr * 16 + rb;
#pragma unroll
      for (int i = 0; i < 8; ++i) dst[(size_t)(r0 + i) * 128 + cc] = acc[i];
    }
    __syncthreads();   // all waves done with W1f/W2f before updating them

    float momv = mom[(size_t)bh * kNC + t];
    float dgv  = 1.f - decay[(size_t)bh * kNC + t];
    const bf16* s1 = g1buf + ((size_t)bh * kNC + t) * 16384;  // col-major, matches state
    const bf16* s2 = g2buf + ((size_t)bh * kNC + t) * 16384;
    for (int i = tid; i < 16384; i += 256) {
      float m1 = momv * M1f[i] + (float)s1[i];
      M1f[i] = m1;
      W1f[i] = dgv * W1f[i] + m1;
      float m2 = momv * M2f[i] + (float)s2[i];
      M2f[i] = m2;
      W2f[i] = dgv * W2f[i] + m2;
    }
    __syncthreads();
  }
}

// ---------------------------------------------------------------
// Per-(b,h,n) rmsnorm over d, *(gamma+1), *gate -> act bf16 (b,n,h*d).
// ---------------------------------------------------------------
__global__ void nm_norm_gate(const float* __restrict__ predbuf, const float* __restrict__ gamma,
                             const float* __restrict__ gate, bf16* __restrict__ act) {
  int tid = threadIdx.x, wv = tid >> 5, l = tid & 31;
  int row = blockIdx.x * 8 + wv;                   // [0, B*H*N)
  int b = row >> 14, h = (row >> 12) & 3, n = row & 4095;
  const float* p = predbuf + (size_t)row * 128;
  float x[4]; float ss = 0.f;
#pragma unroll
  for (int i = 0; i < 4; ++i) { x[i] = p[l + 32 * i]; ss += x[i] * x[i]; }
  ss = wave_sum(ss);
  float rms = rsqrtf(ss * (1.f / 128.f) + 1e-6f);
  float gv = gate[((size_t)b * kN + n) * kH + h];
#pragma unroll
  for (int i = 0; i < 4; ++i) {
    int d = l + 32 * i;
    float o = x[i] * rms * (gamma[h * 128 + d] + 1.f) * gv;
    act[((size_t)b * kN + n) * 512 + h * 128 + d] = (bf16)o;
  }
}

// ---------------------------------------------------------------
// Final GEMM: out(16384x512)f32 = act bf16 @ W_o bf16 (W_o given transposed).
// ---------------------------------------------------------------
__global__ void nm_outgemm(const bf16* __restrict__ X, const bf16* __restrict__ Wt,
                           float* __restrict__ out) {
  __shared__ bf16 Xs[64 * 64];
  __shared__ bf16 Wst[64 * 64];
  int tid = threadIdx.x, wv = tid >> 5, l = tid & 31;
  int row0 = blockIdx.x * 64, col0 = blockIdx.y * 64;
  int tr0 = wv >> 2, tc0 = wv & 3, tr1 = (wv + 8) >> 2, tc1 = (wv + 8) & 3;
  v8f acc0 = {}, acc1 = {};

  for (int kk = 0; kk < 512; kk += 64) {
    for (int i = tid; i < 512; i += 256) {
      int r = i >> 3, c = i & 7;
      ((uint4*)(Xs + r * 64))[c]  = ((const uint4*)(X  + ((size_t)(row0 + r)) * 512 + kk))[c];
      ((uint4*)(Wst + r * 64))[c] = ((const uint4*)(Wt + ((size_t)(col0 + r)) * 512 + kk))[c];
    }
    __syncthreads();
#pragma unroll
    for (int ks = 0; ks < 64; ks += 32) {
      acc0 = wmma_bf16(ld_a_rm(Xs + tr0 * 16 * 64 + ks, 64),
                       ld_b_ct(Wst + tc0 * 16 * 64 + ks, 64), acc0);
      acc1 = wmma_bf16(ld_a_rm(Xs + tr1 * 16 * 64 + ks, 64),
                       ld_b_ct(Wst + tc1 * 16 * 64 + ks, 64), acc1);
    }
    __syncthreads();
  }
  int colL = l & 15, rb = (l >> 4) * 8;
#pragma unroll
  for (int i = 0; i < 8; ++i) {
    int R0 = row0 + tr0 * 16 + rb + i, C0 = col0 + tc0 * 16 + colL;
    out[(size_t)R0 * 512 + C0] = acc0[i];
    int R1 = row0 + tr1 * 16 + rb + i, C1 = col0 + tc1 * 16 + colL;
    out[(size_t)R1 * 512 + C1] = acc1[i];
  }
}

// ---------------------------------------------------------------
extern "C" void kernel_launch(void* const* d_in, const int* in_sizes, int n_in,
                              void* d_out, int out_size, void* d_ws, size_t ws_size,
                              hipStream_t stream) {
  (void)in_sizes; (void)n_in; (void)out_size; (void)ws_size;
  const float* seq    = (const float*)d_in[0];
  const float* gs     = (const float*)d_in[1];
  const float* gr     = (const float*)d_in[2];
  const float* Wq     = (const float*)d_in[3];
  const float* Wkv    = (const float*)d_in[4];
  const float* Wstep  = (const float*)d_in[5];
  const float* Wmom   = (const float*)d_in[6];
  const float* Wdecay = (const float*)d_in[7];
  const float* Wgate  = (const float*)d_in[8];
  const float* gamma  = (const float*)d_in[9];
  const float* Wo     = (const float*)d_in[10];
  const float* MW1    = (const float*)d_in[11];
  const float* MW2    = (const float*)d_in[12];

  char* wsp = (char*)d_ws;
  auto alloc = [&](size_t bytes) { char* p = wsp; wsp += (bytes + 255) & ~(size_t)255; return p; };

  bf16*  s_bf   = (bf16*)alloc((size_t)kB * kN * kDIM * 2);        // 16 MB
  bf16*  r_bf   = (bf16*)alloc((size_t)kB * kN * kDIM * 2);        // 16 MB
  bf16*  k_bf   = (bf16*)alloc((size_t)kBH * kNC * 64 * 128 * 2);  // 16 MB
  bf16*  v_bf   = (bf16*)alloc((size_t)kBH * kNC * 64 * 128 * 2);  // 16 MB
  bf16*  q_bf   = (bf16*)alloc((size_t)kBH * kNC * 64 * 128 * 2);  // 16 MB
  bf16*  g1_bf  = (bf16*)alloc((size_t)kBH * kNC * 128 * 128 * 2); // 32 MB
  bf16*  g2_bf  = (bf16*)alloc((size_t)kBH * kNC * 128 * 128 * 2); // 32 MB
  float* predb  = (float*)alloc((size_t)kBH * kN * 128 * 4);       // 32 MB
  bf16*  act_bf = (bf16*)alloc((size_t)kB * kN * kDIM * 2);        // 16 MB
  float* lrb    = (float*)alloc((size_t)kBH * kNC * 64 * 4);
  float* decayb = (float*)alloc((size_t)kBH * kNC * 4);
  float* momb   = (float*)alloc((size_t)kBH * kNC * 4);
  float* gateb  = (float*)alloc((size_t)kB * kN * kH * 4);
  bf16*  Wq_t   = (bf16*)alloc((size_t)512 * 512 * 2);
  bf16*  Wkv_t  = (bf16*)alloc((size_t)512 * 1024 * 2);
  bf16*  Wo_t   = (bf16*)alloc((size_t)512 * 512 * 2);
  bf16*  MW1_t  = (bf16*)alloc((size_t)128 * 128 * 2);
  bf16*  MW2_t  = (bf16*)alloc((size_t)128 * 128 * 2);
  bf16*  MW2_rm = (bf16*)alloc((size_t)128 * 128 * 2);

  // 0) weight conversions (transposed for B-operand column-major layout)
  nm_f32_to_bf16_t<<<(512 * 512 + 255) / 256, 256, 0, stream>>>(Wq, Wq_t, 512, 512);
  nm_f32_to_bf16_t<<<(512 * 1024 + 255) / 256, 256, 0, stream>>>(Wkv, Wkv_t, 512, 1024);
  nm_f32_to_bf16_t<<<(512 * 512 + 255) / 256, 256, 0, stream>>>(Wo, Wo_t, 512, 512);
  nm_f32_to_bf16_t<<<(128 * 128 + 255) / 256, 256, 0, stream>>>(MW1, MW1_t, 128, 128);
  nm_f32_to_bf16_t<<<(128 * 128 + 255) / 256, 256, 0, stream>>>(MW2, MW2_t, 128, 128);
  nm_f32_to_bf16<<<(128 * 128 + 255) / 256, 256, 0, stream>>>(MW2, MW2_rm, 128 * 128);

  // 1) prep
  nm_prep<<<kB * kNC, 256, 0, stream>>>(seq, gs, gr, Wstep, Wgate, Wdecay, Wmom,
                                        s_bf, r_bf, lrb, gateb, decayb, momb);

  // 2) projections (WMMA)
  nm_proj<<<dim3(256, 16), 256, 0, stream>>>(s_bf, Wkv_t, 1024, k_bf, v_bf);
  nm_proj<<<dim3(256, 8), 256, 0, stream>>>(r_bf, Wq_t, 512, q_bf, nullptr);

  // 3) per-chunk gradients (WMMA), 256KB dynamic LDS
  size_t grad_lds = 32768 + 8 * 16384 + 3 * 32768; // 262144
  nm_grad<<<dim3(kNC, kBH), 256, grad_lds, stream>>>(k_bf, v_bf, MW1_t, MW2_t, MW2_rm,
                                                     lrb, g1_bf, g2_bf);

  // 4) sequential scan + retrieval (WMMA), 288KB dynamic LDS
  size_t scan_lds = 4 * 65536 + 2 * 16384; // 294912
  nm_scan<<<kBH, 256, scan_lds, stream>>>(MW1, MW2, q_bf, g1_bf, g2_bf, momb, decayb, predb);

  // 5) rmsnorm / gamma / gate
  nm_norm_gate<<<(kBH * kN) / 8, 256, 0, stream>>>(predb, gamma, gateb, act_bf);

  // 6) final output GEMM (WMMA)
  nm_outgemm<<<dim3(256, 8), 256, 0, stream>>>(act_bf, Wo_t, (float*)d_out);
}
